// MoE_68547678044793
// MI455X (gfx1250) — matrix operand order repaired and checked by reading, not compiled, not run on hardware
//
#include <hip/hip_runtime.h>
#include <hip/hip_bf16.h>

// ---------------------------------------------------------------------------
// MoE (top-2 of 8 experts + 1 shared) for MI455X / gfx1250, wave32 + WMMA.
// bf16 WMMA (v_wmma_f32_16x16x32_bf16) with fp32 accumulation; deterministic
// token->expert grouping; per-expert kernels serialized on the stream so the
// y accumulation order is fixed (no atomics).
// A-tiles staged via CDNA5 async global->LDS (ASYNCcnt); B-tiles converted
// fp32->bf16 on the fly with paired-k b32 LDS stores + XOR chunk swizzle.
// ---------------------------------------------------------------------------

#define TTOK 4096   // B*S
#define DEMB 1024
#define DFF  4096
#define NEXP 8

#define BM 128      // token-tile per block
#define BN 128      // column-tile per block
#define BK 32       // K-step (bf16 WMMA K)
#define LDT 40      // LDS stride (elems) for bf16 tiles: 80B rows, 16B-aligned
#define LDE 132     // LDS stride for fp32 epilogue tile

typedef __attribute__((ext_vector_type(16))) __bf16 v16bf;
typedef __attribute__((ext_vector_type(8)))  float  v8f;
typedef __attribute__((ext_vector_type(4)))  unsigned int u32x4;
typedef __attribute__((ext_vector_type(4)))  float  f32x4;

union Frag { v16bf v; u32x4 q[2]; };

static __device__ __forceinline__ unsigned short f2bf(float f) {
  unsigned int u = __float_as_uint(f);
  u += 0x7FFFu + ((u >> 16) & 1u);   // round-to-nearest-even
  return (unsigned short)(u >> 16);
}

static __device__ __forceinline__ unsigned pack_bf2(float lo, float hi) {
  return (unsigned)f2bf(lo) | ((unsigned)f2bf(hi) << 16);
}

static __device__ __forceinline__ float gelu_exact(float v) {
  return 0.5f * v * (1.0f + erff(v * 0.70710678118654752f));
}

// CDNA5 async copy: 16B global -> LDS, tracked by ASYNCcnt (no VGPR data).
// lds_off is the wave-relative LDS byte address (low 32 bits of the generic
// address of a __shared__ object are the LDS offset on AMDGPU).
static __device__ __forceinline__ void async_ld_b128(unsigned lds_off,
                                                     const void* gaddr) {
  asm volatile("global_load_async_to_lds_b128 %0, %1, off"
               :: "v"(lds_off), "v"(gaddr) : "memory");
}
static __device__ __forceinline__ void wait_async0() {
  asm volatile("s_wait_asynccnt 0x0" ::: "memory");
}

static __device__ __forceinline__ unsigned lds_off_of(const void* p) {
  return (unsigned)(unsigned long long)p;
}

// Bt byte offset with 16B-chunk XOR swizzle keyed on row>>4:
// staging stores (4 consecutive rows per lane) drop to 2-way conflicts,
// fragment loads stay conflict-free and 16B-aligned.
static __device__ __forceinline__ int bt_off(int n, int kByte) {
  int c = (kByte >> 4) ^ ((n >> 4) & 3);
  return n * (LDT * 2) + (c << 4) + (kByte & 15);
}

// --------------------------- init: zero y, x -> bf16 ------------------------
__global__ __launch_bounds__(256) void moe_init_kernel(
    const float* __restrict__ x, unsigned short* __restrict__ xb,
    float* __restrict__ y) {
  int i = blockIdx.x * 256 + threadIdx.x;
  const int stride = 4096 * 256;          // grid is 4096 blocks
  #pragma unroll
  for (int k = 0; k < 4; ++k) {           // covers TTOK*DEMB = 4M elements
    int idx = i + k * stride;
    xb[idx] = f2bf(x[idx]);
    y[idx] = 0.0f;
  }
}

// --------------------------- router: gate[T, 8] -----------------------------
__global__ __launch_bounds__(256) void moe_router_kernel(
    const float* __restrict__ x, const float* __restrict__ Wg,
    const float* __restrict__ bg, float* __restrict__ gate) {
  int lane = threadIdx.x & 31;
  int w = threadIdx.x >> 5;
  int t = blockIdx.x * 8 + w;
  float acc[NEXP];
  #pragma unroll
  for (int e = 0; e < NEXP; ++e) acc[e] = 0.0f;
  const float* xr = x + (size_t)t * DEMB;
  for (int d = lane; d < DEMB; d += 32) {
    float xv = xr[d];
    const f32x4* wr = (const f32x4*)(Wg + d * NEXP);
    f32x4 a = wr[0], b = wr[1];
    acc[0] += xv * a.x; acc[1] += xv * a.y; acc[2] += xv * a.z; acc[3] += xv * a.w;
    acc[4] += xv * b.x; acc[5] += xv * b.y; acc[6] += xv * b.z; acc[7] += xv * b.w;
  }
  #pragma unroll
  for (int off = 16; off > 0; off >>= 1) {
    #pragma unroll
    for (int e = 0; e < NEXP; ++e) acc[e] += __shfl_xor(acc[e], off, 32);
  }
  if (lane == 0) {
    float p[NEXP];
    float mx = -3.0e38f;
    #pragma unroll
    for (int e = 0; e < NEXP; ++e) { p[e] = acc[e] + bg[e]; mx = fmaxf(mx, p[e]); }
    float s = 0.0f;
    #pragma unroll
    for (int e = 0; e < NEXP; ++e) { p[e] = __expf(p[e] - mx); s += p[e]; }
    float inv = 1.0f / s;
    int i1 = 0;
    #pragma unroll
    for (int e = 1; e < NEXP; ++e) if (p[e] > p[i1]) i1 = e;   // lowest-index tie
    int i2 = (i1 == 0) ? 1 : 0;
    #pragma unroll
    for (int e = 0; e < NEXP; ++e) if (e != i1 && p[e] > p[i2]) i2 = e;
    #pragma unroll
    for (int e = 0; e < NEXP; ++e)
      gate[t * NEXP + e] = (e == i1 || e == i2) ? p[e] * inv : 0.0f;
  }
}

// ----------------- deterministic stable per-expert token lists --------------
__global__ __launch_bounds__(256) void moe_lists_kernel(
    const float* __restrict__ gate, int* __restrict__ lists,
    int* __restrict__ cnt) {
  __shared__ int s[256];
  const int tid = threadIdx.x;
  for (int e = 0; e < NEXP; ++e) {
    int flags[TTOK / 256];
    int c = 0;
    #pragma unroll
    for (int i = 0; i < TTOK / 256; ++i) {
      int t = tid * (TTOK / 256) + i;
      flags[i] = (gate[t * NEXP + e] > 0.0f) ? 1 : 0;
      c += flags[i];
    }
    s[tid] = c;
    __syncthreads();
    for (int off = 1; off < 256; off <<= 1) {
      int add = (tid >= off) ? s[tid - off] : 0;
      __syncthreads();
      s[tid] += add;
      __syncthreads();
    }
    int pos = s[tid] - c;  // exclusive prefix -> stable token order
    #pragma unroll
    for (int i = 0; i < TTOK / 256; ++i) {
      if (flags[i]) lists[e * TTOK + pos++] = tid * (TTOK / 256) + i;
    }
    if (tid == 255) cnt[e] = s[255];
    __syncthreads();
  }
}

// --------------------------- grouped GEMM (WMMA) ----------------------------
// PHASE 1: h[r,:] = gelu(xb[list[r],:] @ W1 + b1)   (bf16 out)
// PHASE 2: y[list[r],:] += gate * (h[r,:] @ W2 + b2)
template <int PHASE>
__global__ __launch_bounds__(256) void moe_gemm_kernel(
    const unsigned short* __restrict__ Abf,   // PHASE1: xb ; PHASE2: h
    const float* __restrict__ W,              // fp32 weights (this expert)
    const float* __restrict__ bias,
    const int* __restrict__ list,             // nullptr for shared expert
    const int* __restrict__ cntp,             // nullptr for shared expert
    const float* __restrict__ gate, int eidx, // PHASE2 only
    unsigned short* __restrict__ hOut,        // PHASE1 out
    float* __restrict__ yOut,                 // PHASE2 out (accumulate)
    int is_shared) {
  constexpr int K = (PHASE == 1) ? DEMB : DFF;
  constexpr int N = (PHASE == 1) ? DFF : DEMB;

  __shared__ unsigned short At[BM * LDT];   // linear [m][k], async-filled
  __shared__ unsigned short Bt[BN * LDT];   // [n][k], chunk-swizzled
  __shared__ float epi[BM * LDE];
  __shared__ int rowtok[BM];

  const int count = cntp ? cntp[0] : TTOK;
  const int r0 = (int)blockIdx.y * BM;
  if (r0 >= count) return;                  // uniform early exit
  const int n0 = (int)blockIdx.x * BN;

  const int tid = threadIdx.x;
  const int lane = tid & 31;
  const int w = tid >> 5;
  const int half = lane >> 4;               // 0: lanes 0-15, 1: lanes 16-31
  const int l = lane & 15;

  if (PHASE == 1 && tid < BM) {
    int r = r0 + tid;
    // clamp out-of-range rows to token 0: they compute garbage that the
    // epilogue masks out, so no zero-fill / no divergent loads are needed.
    rowtok[tid] = (r < count) ? (is_shared ? r : list[r]) : 0;
  }
  __syncthreads();

  // ---- hoisted A staging addresses: 2x 16B async copies per thread --------
  const unsigned at_base = lds_off_of(At);
  const unsigned short* aSrc[2];
  unsigned aDst[2];
  #pragma unroll
  for (int i = 0; i < 2; ++i) {
    int lin = tid + i * 256;                // 512 chunks of 16B (8 bf16)
    int m = lin >> 2;                       // 4 chunks per 64B row
    int kq = (lin & 3) << 3;                // element offset 0,8,16,24
    int row = (PHASE == 1) ? rowtok[m] : (r0 + m);
    aSrc[i] = Abf + (size_t)row * K + kq;
    aDst[i] = at_base + (unsigned)((m * LDT + kq) * 2);
  }

  v8f acc[8] = {};

  for (int k0 = 0; k0 < K; k0 += BK) {
    // ---- A tile: async global -> LDS, no VGPR round trip ----
    async_ld_b128(aDst[0], aSrc[0] + k0);
    async_ld_b128(aDst[1], aSrc[1] + k0);

    // ---- B tile: 2 consecutive K rows per thread, pack (k,k+1) -> b32 ----
    #pragma unroll
    for (int i = 0; i < 2; ++i) {
      int lin = tid + i * 256;              // 512 units
      int k = (lin >> 5) << 1;              // even k: 0..30
      int nq = (lin & 31) << 2;             // 0..124
      const float* wp = W + (size_t)(k0 + k) * N + n0 + nq;
      f32x4 ra = *(const f32x4*)(wp);
      f32x4 rb = *(const f32x4*)(wp + N);
      char* bb = (char*)Bt;
      *(unsigned*)(bb + bt_off(nq + 0, 2 * k)) = pack_bf2(ra.x, rb.x);
      *(unsigned*)(bb + bt_off(nq + 1, 2 * k)) = pack_bf2(ra.y, rb.y);
      *(unsigned*)(bb + bt_off(nq + 2, 2 * k)) = pack_bf2(ra.z, rb.z);
      *(unsigned*)(bb + bt_off(nq + 3, 2 * k)) = pack_bf2(ra.w, rb.w);
    }
    wait_async0();
    __syncthreads();

    // A frag: lanes 0-15 -> K{0..7,16..23}; lanes 16-31 -> K{8..15,24..31}
    Frag a;
    const unsigned short* ap = &At[(w * 16 + l) * LDT];
    a.q[0] = *(const u32x4*)(ap + half * 8);
    a.q[1] = *(const u32x4*)(ap + 16 + half * 8);

    // prefetch all 8 B frags, then run the WMMA chain off one dscnt wait
    Frag bf[8];
    #pragma unroll
    for (int nt = 0; nt < 8; ++nt) {
      const char* bb = (const char*)Bt;
      int n = nt * 16 + l;
      bf[nt].q[0] = *(const u32x4*)(bb + bt_off(n, half * 16));
      bf[nt].q[1] = *(const u32x4*)(bb + bt_off(n, 32 + half * 16));
    }
    #pragma unroll
    for (int nt = 0; nt < 8; ++nt) {
      acc[nt] = __builtin_amdgcn_wmma_f32_16x16x32_bf16(
          false, a.v, false, bf[nt].v, (short)0, acc[nt], false, false);
    }
    __syncthreads();
  }

  // ---- epilogue: restage fp32 accs through LDS for coalesced stores ----
  #pragma unroll
  for (int nt = 0; nt < 8; ++nt) {
    #pragma unroll
    for (int r = 0; r < 8; ++r)
      epi[(w * 16 + half * 8 + r) * LDE + nt * 16 + l] = acc[nt][r];
  }
  __syncthreads();

  int row = tid >> 1;                 // 2 threads per row, 64 cols each
  int cb = (tid & 1) * 64;
  int r = r0 + row;
  if (r < count) {
    if (PHASE == 1) {
      size_t base = (size_t)r * N + n0;
      for (int c = cb; c < cb + 64; c += 2) {
        float v0 = gelu_exact(epi[row * LDE + c]     + bias[n0 + c]);
        float v1 = gelu_exact(epi[row * LDE + c + 1] + bias[n0 + c + 1]);
        *(unsigned int*)(&hOut[base + c]) = pack_bf2(v0, v1);
      }
    } else {
      int tok = is_shared ? r : list[r];
      float g = is_shared ? 1.0f : gate[tok * NEXP + eidx];
      size_t base = (size_t)tok * N + n0;
      for (int c = cb; c < cb + 64; ++c)
        yOut[base + c] += g * (epi[row * LDE + c] + bias[n0 + c]);
    }
  }
}

// ---------------------------------------------------------------------------
extern "C" void kernel_launch(void* const* d_in, const int* in_sizes, int n_in,
                              void* d_out, int out_size, void* d_ws,
                              size_t ws_size, hipStream_t stream) {
  const float* x   = (const float*)d_in[0];
  const float* Wg  = (const float*)d_in[1];
  const float* bg  = (const float*)d_in[2];
  const float* W1  = (const float*)d_in[3];
  const float* b1  = (const float*)d_in[4];
  const float* W2  = (const float*)d_in[5];
  const float* b2  = (const float*)d_in[6];
  const float* Ws1 = (const float*)d_in[7];
  const float* bs1 = (const float*)d_in[8];
  const float* Ws2 = (const float*)d_in[9];
  const float* bs2 = (const float*)d_in[10];
  float* y = (float*)d_out;

  // workspace layout (~40.3 MB total)
  char* ws = (char*)d_ws;
  unsigned short* xb = (unsigned short*)ws;                    // 8 MB
  size_t off = (size_t)TTOK * DEMB * 2;
  float* gate = (float*)(ws + off);  off += (size_t)TTOK * NEXP * 4;  // 128 KB
  int* lists  = (int*)(ws + off);    off += (size_t)NEXP * TTOK * 4;  // 128 KB
  int* cnt    = (int*)(ws + off);    off += 256;
  unsigned short* h = (unsigned short*)(ws + off);             // 32 MB
  (void)in_sizes; (void)n_in; (void)out_size; (void)ws_size;

  moe_init_kernel<<<4096, 256, 0, stream>>>(x, xb, y);
  moe_router_kernel<<<TTOK / 8, 256, 0, stream>>>(x, Wg, bg, gate);
  moe_lists_kernel<<<1, 256, 0, stream>>>(gate, lists, cnt);

  dim3 g1(DFF / BN, TTOK / BM);
  dim3 g2(DEMB / BN, TTOK / BM);
  for (int e = 0; e < NEXP; ++e) {
    moe_gemm_kernel<1><<<g1, 256, 0, stream>>>(
        xb, W1 + (size_t)e * DEMB * DFF, b1 + (size_t)e * DFF,
        lists + (size_t)e * TTOK, cnt + e, nullptr, 0, h, nullptr, 0);
    moe_gemm_kernel<2><<<g2, 256, 0, stream>>>(
        h, W2 + (size_t)e * DFF * DEMB, b2 + (size_t)e * DEMB,
        lists + (size_t)e * TTOK, cnt + e, gate, e, nullptr, y, 0);
  }
  // shared expert (N_SHARED == 1, so the 1/ns mean is a no-op)
  moe_gemm_kernel<1><<<g1, 256, 0, stream>>>(
      xb, Ws1, bs1, nullptr, nullptr, nullptr, 0, h, nullptr, 1);
  moe_gemm_kernel<2><<<g2, 256, 0, stream>>>(
      h, Ws2, bs2, nullptr, nullptr, nullptr, 0, nullptr, y, 1);
}
